// RayTracer_23948737642615
// MI455X (gfx1250) — compile-verified, matrix operand order checked by reading, not run on hardware
//
#include <hip/hip_runtime.h>
#include <hip/hip_bf16.h>
#include <math.h>

// ---------------------------------------------------------------------------
// Neural-SDF ray tracer for MI455X (gfx1250, wave32, WMMA + TDM).
// Hot loop: batched MLP via v_wmma_f32_16x16x32_f16 (16 points x 64 hidden).
// Weight staging: tensor_load_to_lds (TDM) per-wave 1 KB slices.
// ---------------------------------------------------------------------------

typedef __attribute__((ext_vector_type(16))) _Float16 v16h;
typedef __attribute__((ext_vector_type(8)))  float    v8f;

#define N_RAYS   8192
#define N_STEPS  256
#define HID      64
#define SDF_THR  5e-05f
#define WAVES    8          // 256 threads / wave32

// Branch-free tanh: tanh(x) = 1 - 2/(exp2(2x*log2e)+1).
// Saturates correctly for |x| large (exp2 -> inf or 0). No exec-mask churn.
__device__ __forceinline__ float fast_tanh(float x) {
#if defined(__AMDGCN__)
  const float e = __builtin_amdgcn_exp2f(x * 2.8853900817779268f); // 2*log2(e)
  return 1.0f - 2.0f * __builtin_amdgcn_rcpf(e + 1.0f);
#else
  return tanhf(x);
#endif
}

// Shared-memory staging for the MLP.
struct MlpShared {
  __align__(32) _Float16 w2frag[2][4][32][16];  // [ktile][ntile][lane][half] 8 KB
  __align__(16) _Float16 hbuf[WAVES][32][64];   // per-wave h1 then h2      32 KB
  float w1[3][HID];
  float b1v[HID];
  float b2v[HID];
  float w3v[HID];
  float b3v;
};

__device__ __forceinline__ void stage_weights(MlpShared& sh, const _Float16* frag,
                                              const float* W1, const float* b1,
                                              const float* b2, const float* W3,
                                              const float* b3) {
  const int t = threadIdx.x;
#if defined(__gfx1250__)
  // ---- Tensor Data Mover: each wave DMAs its 1 KB slice of the 8 KB W2
  // fragment buffer straight into LDS (tracked by TENSORcnt, EXEC ignored).
  {
    __builtin_prefetch(frag, 0, 3);  // global_prefetch_b8 (speculative, RT)
    const int w = __builtin_amdgcn_readfirstlane((int)(threadIdx.x >> 5));
    const unsigned long long gaddr =
        (unsigned long long)(uintptr_t)frag + (unsigned long long)w * 1024ull;
    const unsigned int laddr =
        (unsigned int)(uintptr_t)(&sh.w2frag[0][0][0][0]) + (unsigned int)w * 1024u;

    typedef __attribute__((ext_vector_type(4))) unsigned int u32x4;
    typedef __attribute__((ext_vector_type(8))) unsigned int u32x8;
    union { u32x4 v; unsigned int d[4]; } g0;
    union { u32x8 v; unsigned int d[8]; } g1;
    // D# group 0: count=1, lds_addr, global_addr[56:0], type=2 ("image")
    g0.d[0] = 1u;
    g0.d[1] = laddr;
    g0.d[2] = (unsigned int)gaddr;
    g0.d[3] = (unsigned int)((gaddr >> 32) & 0x01FFFFFFu) | 0x80000000u;
    // D# group 1: data_size=2 (4B), 1-D tensor/tile of 256 elements
    g1.d[0] = 0x00020000u;        // workgroup_mask=0, data_size=4B, no flags
    g1.d[1] = 256u << 16;         // tensor_dim0 = 256 (bits 63:48)
    g1.d[2] = 1u   << 16;         // tensor_dim1 = 1   (bits 95:80)
    g1.d[3] = 256u << 16;         // tile_dim0   = 256 (bits 127:112)
    g1.d[4] = 1u;                 // tile_dim1   = 1   (bits 143:128)
    g1.d[5] = 256u;               // tensor_dim0_stride = 256
    g1.d[6] = 0u;
    g1.d[7] = 0u;
    asm volatile("tensor_load_to_lds %0, %1" :: "s"(g0.v), "s"(g1.v) : "memory");
    __builtin_amdgcn_s_wait_tensorcnt(0);
  }
#else
  // Fallback (host pass / non-gfx1250): plain vector copy.
  {
    const uint4* src = reinterpret_cast<const uint4*>(frag);
    uint4*       dst = reinterpret_cast<uint4*>(&sh.w2frag[0][0][0][0]);
    for (int i = t; i < 512; i += 256) dst[i] = src[i];
  }
#endif
  for (int i = t; i < 3 * HID; i += 256) sh.w1[i / HID][i % HID] = W1[i];
  if (t < HID) { sh.b1v[t] = b1[t]; sh.b2v[t] = b2[t]; sh.w3v[t] = W3[t]; }
  if (t == 0) sh.b3v = b3[0];
  __syncthreads();
}

// Wave-cooperative SDF eval: each lane owns one point (32 points / wave).
// Must be called by full, non-divergent 256-thread blocks (EXEC all ones).
__device__ float sdf_eval_wave(float px, float py, float pz, MlpShared& sh) {
  const int t    = threadIdx.x;
  const int wave = t >> 5;
  const int lane = t & 31;
  const int l15  = lane & 15;
  const int kb   = (lane >= 16) ? 8 : 0;   // A/B fragment K sub-block per lane half

  const float base = sqrtf(px * px + py * py + pz * pz) - 1.0f;

  // ---- layer 1 (K=3, VALU) -> h1 row for this lane's point ----
#pragma unroll 8
  for (int n = 0; n < HID; ++n) {
    float a = fmaf(px, sh.w1[0][n],
             fmaf(py, sh.w1[1][n],
             fmaf(pz, sh.w1[2][n], sh.b1v[n])));
    sh.hbuf[wave][lane][n] = (_Float16)fast_tanh(a);
  }
  __syncthreads();

  // ---- gather A fragments (2 point-tiles x 2 K-halves), ISA 16-bit A layout ----
  union HU { v16h v; _Float16 h[16]; };
  HU afrag[2][2];
#pragma unroll
  for (int pt = 0; pt < 2; ++pt) {
    const int m = pt * 16 + l15;
#pragma unroll
    for (int kh = 0; kh < 2; ++kh) {
      const _Float16* row = &sh.hbuf[wave][m][kh * 32];
#pragma unroll
      for (int j = 0; j < 8; ++j) {
        afrag[pt][kh].h[j]     = row[kb + j];        // K = kb..kb+7
        afrag[pt][kh].h[8 + j] = row[16 + kb + j];   // K = 16+kb..16+kb+7
      }
    }
  }
  __syncthreads();

  // ---- layer 2: 16 x v_wmma_f32_16x16x32_f16 per wave; h2 overwrites hbuf ----
  const int mb = (lane >= 16) ? 8 : 0;               // C/D layout M offset
#pragma unroll
  for (int pt = 0; pt < 2; ++pt) {
#pragma unroll
    for (int nt = 0; nt < 4; ++nt) {
      v16h b0  = *reinterpret_cast<const v16h*>(&sh.w2frag[0][nt][lane][0]);
      v16h b1f = *reinterpret_cast<const v16h*>(&sh.w2frag[1][nt][lane][0]);
      v8f c = {};
      c = __builtin_amdgcn_wmma_f32_16x16x32_f16(false, afrag[pt][0].v,
                                                 false, b0, (short)0, c, false, false);
      c = __builtin_amdgcn_wmma_f32_16x16x32_f16(false, afrag[pt][1].v,
                                                 false, b1f, (short)0, c, false, false);
      union FU { v8f v; float f[8]; } cu; cu.v = c;
      const float bias = sh.b2v[nt * 16 + l15];
#pragma unroll
      for (int r = 0; r < 8; ++r) {
        float h2 = fast_tanh(cu.f[r] + bias);
        sh.hbuf[wave][pt * 16 + mb + r][nt * 16 + l15] = (_Float16)h2;
      }
    }
  }
  __syncthreads();

  // ---- layer 3 (64 -> 1, VALU) ----
  float acc = sh.b3v;
#pragma unroll 8
  for (int k = 0; k < HID; ++k)
    acc = fmaf((float)sh.hbuf[wave][lane][k], sh.w3v[k], acc);
  __syncthreads();

  return base + 0.05f * acc;
}

// ---------------------------------------------------------------------------
// Kernels
// ---------------------------------------------------------------------------

// Pre-swizzle W2 (f32 row-major 64x64) into WMMA B-fragment layout, f16.
__global__ __launch_bounds__(256) void k_prep(const float* __restrict__ W2,
                                              _Float16* __restrict__ frag) {
  for (int e = threadIdx.x; e < 4096; e += 256) {
    const int j    = e & 15;
    const int lane = (e >> 4) & 31;
    const int nt   = (e >> 9) & 3;
    const int kt   = (e >> 11) & 1;
    const int n = nt * 16 + (lane & 15);
    const int K = kt * 32 + ((j >> 3) * 16) + ((lane >= 16) ? 8 : 0) + (j & 7);
    frag[e] = (_Float16)W2[K * HID + n];
  }
}

// Evaluate sdf at p = o + d * tval[r]; write out[r]. 32 blocks x 256 threads.
__global__ __launch_bounds__(256) void k_sdf_at(
    const float* __restrict__ ray_o, const float* __restrict__ ray_d,
    const float* __restrict__ tval, const _Float16* __restrict__ frag,
    const float* __restrict__ W1, const float* __restrict__ b1,
    const float* __restrict__ b2, const float* __restrict__ W3,
    const float* __restrict__ b3, float* __restrict__ out) {
  __shared__ MlpShared sh;
  stage_weights(sh, frag, W1, b1, b2, W3, b3);
  const int r = blockIdx.x * 256 + threadIdx.x;
  const float dm = tval[r];
  const float f = sdf_eval_wave(ray_o[3*r+0] + ray_d[3*r+0] * dm,
                                ray_o[3*r+1] + ray_d[3*r+1] * dm,
                                ray_o[3*r+2] + ray_d[3*r+2] * dm, sh);
  out[r] = f;
}

// One bisection step on all rays.
__global__ __launch_bounds__(256) void k_bisect(
    const float* __restrict__ ray_o, const float* __restrict__ ray_d,
    float* __restrict__ dlow, float* __restrict__ dhigh,
    const _Float16* __restrict__ frag,
    const float* __restrict__ W1, const float* __restrict__ b1,
    const float* __restrict__ b2, const float* __restrict__ W3,
    const float* __restrict__ b3) {
  __shared__ MlpShared sh;
  stage_weights(sh, frag, W1, b1, b2, W3, b3);
  const int r = blockIdx.x * 256 + threadIdx.x;
  const float dl = dlow[r], dh = dhigh[r];
  const float dm = 0.5f * (dl + dh);
  const float f = sdf_eval_wave(ray_o[3*r+0] + ray_d[3*r+0] * dm,
                                ray_o[3*r+1] + ray_d[3*r+1] * dm,
                                ray_o[3*r+2] + ray_d[3*r+2] * dm, sh);
  const bool pos = f > 0.0f;
  dlow[r]  = pos ? dm : dl;
  dhigh[r] = pos ? dh : dm;
}

// Final midpoint eval: d_mid = 0.5*(dl+dh), f_mid = sdf(d_mid).
__global__ __launch_bounds__(256) void k_mid(
    const float* __restrict__ ray_o, const float* __restrict__ ray_d,
    const float* __restrict__ dlow, const float* __restrict__ dhigh,
    const _Float16* __restrict__ frag,
    const float* __restrict__ W1, const float* __restrict__ b1,
    const float* __restrict__ b2, const float* __restrict__ W3,
    const float* __restrict__ b3,
    float* __restrict__ dmid, float* __restrict__ fmid) {
  __shared__ MlpShared sh;
  stage_weights(sh, frag, W1, b1, b2, W3, b3);
  const int r = blockIdx.x * 256 + threadIdx.x;
  const float dm = 0.5f * (dlow[r] + dhigh[r]);
  const float f = sdf_eval_wave(ray_o[3*r+0] + ray_d[3*r+0] * dm,
                                ray_o[3*r+1] + ray_d[3*r+1] * dm,
                                ray_o[3*r+2] + ray_d[3*r+2] * dm, sh);
  dmid[r] = dm;
  fmid[r] = f;
}

// Fused dense sampler + per-ray reductions. One block per ray (256 steps).
__global__ __launch_bounds__(256) void k_sample(
    const float* __restrict__ ray_o, const float* __restrict__ ray_d,
    const float* __restrict__ min_dis, const float* __restrict__ max_dis,
    const _Float16* __restrict__ frag,
    const float* __restrict__ W1, const float* __restrict__ b1,
    const float* __restrict__ b2, const float* __restrict__ W3,
    const float* __restrict__ b3,
    float* __restrict__ ssdf, float* __restrict__ sdis, float* __restrict__ rff,
    float* __restrict__ dlow, float* __restrict__ dhigh) {
  __shared__ MlpShared sh;
  __shared__ float sv_arr[256];
  __shared__ float red_val[256];
  __shared__ int   red_idx[256];
  __shared__ int   fneg[256];
  stage_weights(sh, frag, W1, b1, b2, W3, b3);

  const int r = blockIdx.x;
  const int s = threadIdx.x;
  const float tmn = min_dis[r], tmx = max_dis[r];
  const float dis = fmaf((float)s * (1.0f / 255.0f), tmx - tmn, tmn);
  const float sv = sdf_eval_wave(ray_o[3*r+0] + ray_d[3*r+0] * dis,
                                 ray_o[3*r+1] + ray_d[3*r+1] * dis,
                                 ray_o[3*r+2] + ray_d[3*r+2] * dis, sh);
  sv_arr[s]  = sv;
  red_val[s] = fabsf(sv);
  red_idx[s] = s;
  fneg[s]    = (sv < 0.0f) ? s : 0x7fffffff;   // first sign transition
  __syncthreads();
  for (int off = 128; off > 0; off >>= 1) {
    if (s < off) {
      const float v2 = red_val[s + off];
      const int   i2 = red_idx[s + off];
      if (v2 < red_val[s] || (v2 == red_val[s] && i2 < red_idx[s])) {
        red_val[s] = v2; red_idx[s] = i2;
      }
      const int f2 = fneg[s + off];
      if (f2 < fneg[s]) fneg[s] = f2;
    }
    __syncthreads();
  }
  if (s == 0) {
    const int amin = red_idx[0];
    ssdf[r] = sv_arr[amin];
    sdis[r] = fmaf((float)amin * (1.0f / 255.0f), tmx - tmn, tmn);
    const int fn  = fneg[0];
    const int rfm = (fn != 0x7fffffff && fn >= 1) ? 1 : 0;
    int idx = (fn == 0x7fffffff) ? 1 : (fn < 1 ? 1 : fn);
    dlow[r]  = fmaf((float)(idx - 1) * (1.0f / 255.0f), tmx - tmn, tmn);
    dhigh[r] = fmaf((float)idx       * (1.0f / 255.0f), tmx - tmn, tmn);
    rff[r] = (float)rfm;
  }
}

// Merge sampler + rootfind + sphere-trace + mock floor; write 6 outputs.
__global__ __launch_bounds__(256) void k_final(
    const float* __restrict__ ray_o, const float* __restrict__ ray_d,
    const float* __restrict__ min_o, const float* __restrict__ max_o,
    const int* __restrict__ work_mask, const float* __restrict__ floor_dist,
    const float* __restrict__ sdf0a, const float* __restrict__ ssdf,
    const float* __restrict__ sdis, const float* __restrict__ rff,
    const float* __restrict__ dmid, const float* __restrict__ fmid,
    float* __restrict__ out) {
  const int r = blockIdx.x * 256 + threadIdx.x;
  const float sdf0 = sdf0a[r];
  const float acc0 = min_o[r];
  const bool  work   = work_mask[r] != 0;
  const bool  within = acc0 < max_o[r];
  const bool  unfin  = work && (fabsf(sdf0) > SDF_THR) && within;
  bool conv = work && !unfin && (fabsf(sdf0) <= SDF_THR) && within;
  const bool rf = rff[r] > 0.5f;
  const float s_dis = rf ? dmid[r] : sdis[r];
  const float s_sdf = rf ? fmid[r] : ssdf[r];
  conv = unfin ? rf : conv;
  float dist = unfin ? s_dis : acc0;
  float sdfo = unfin ? s_sdf : sdf0;
  const float ox = ray_o[3*r+0], oy = ray_o[3*r+1], oz = ray_o[3*r+2];
  const float dx = ray_d[3*r+0], dy = ray_d[3*r+1], dz = ray_d[3*r+2];
  const float rpx = ox + dx * dist, rpy = oy + dy * dist, rpz = oz + dz * dist;
  const bool realmask = conv;
  float px = rpx, py = rpy, pz = rpz;
  const float fd = floor_dist[r];
  if (!conv) {  // mock floor fill
    px = ox + dx * fd; py = oy + dy * fd; pz = oz + dz * fd;
    sdfo = 0.0f; dist = fd;
  }
  const int Nn = N_RAYS;
  out[r]               = 1.0f;                       // conv (all true after floor)
  out[Nn + r]          = realmask ? 1.0f : 0.0f;     // real_mask
  out[2 * Nn + 3 * r + 0] = px;                      // points
  out[2 * Nn + 3 * r + 1] = py;
  out[2 * Nn + 3 * r + 2] = pz;
  out[5 * Nn + 3 * r + 0] = rpx;                     // real_points (pre-floor)
  out[5 * Nn + 3 * r + 1] = rpy;
  out[5 * Nn + 3 * r + 2] = rpz;
  out[8 * Nn + r]      = sdfo;                       // sdf
  out[9 * Nn + r]      = dist;                       // dist
}

// ---------------------------------------------------------------------------
extern "C" void kernel_launch(void* const* d_in, const int* in_sizes, int n_in,
                              void* d_out, int out_size, void* d_ws, size_t ws_size,
                              hipStream_t stream) {
  const float* ray_o      = (const float*)d_in[0];
  const float* ray_d      = (const float*)d_in[1];
  const float* min_dis    = (const float*)d_in[2];
  const float* max_dis    = (const float*)d_in[3];
  const float* min_o      = (const float*)d_in[4];
  const float* max_o      = (const float*)d_in[5];
  const int*   work_mask  = (const int*)d_in[6];
  const float* floor_dist = (const float*)d_in[7];
  const float* W1 = (const float*)d_in[8];
  const float* b1 = (const float*)d_in[9];
  const float* W2 = (const float*)d_in[10];
  const float* b2 = (const float*)d_in[11];
  const float* W3 = (const float*)d_in[12];
  const float* b3 = (const float*)d_in[13];

  float* ws = (float*)d_ws;
  _Float16* frag = (_Float16*)ws;          // 4096 halves = 2048 floats
  float* sdf0  = ws + 2048;
  float* ssdf  = sdf0  + N_RAYS;
  float* sdis  = ssdf  + N_RAYS;
  float* rff   = sdis  + N_RAYS;
  float* dlow  = rff   + N_RAYS;
  float* dhigh = dlow  + N_RAYS;
  float* dmid  = dhigh + N_RAYS;
  float* fmid  = dmid  + N_RAYS;

  const int RB = N_RAYS / 256;  // 32 blocks for per-ray kernels

  k_prep<<<1, 256, 0, stream>>>(W2, frag);
  // sdf at pts0 = o + d * min_dis_outer
  k_sdf_at<<<RB, 256, 0, stream>>>(ray_o, ray_d, min_o, frag, W1, b1, b2, W3, b3, sdf0);
  // dense sampler + reductions (one block per ray)
  k_sample<<<N_RAYS, 256, 0, stream>>>(ray_o, ray_d, min_dis, max_dis, frag,
                                       W1, b1, b2, W3, b3,
                                       ssdf, sdis, rff, dlow, dhigh);
  // 8 bisection iterations
  for (int i = 0; i < 8; ++i)
    k_bisect<<<RB, 256, 0, stream>>>(ray_o, ray_d, dlow, dhigh, frag,
                                     W1, b1, b2, W3, b3);
  k_mid<<<RB, 256, 0, stream>>>(ray_o, ray_d, dlow, dhigh, frag,
                                W1, b1, b2, W3, b3, dmid, fmid);
  k_final<<<RB, 256, 0, stream>>>(ray_o, ray_d, min_o, max_o, work_mask, floor_dist,
                                  sdf0, ssdf, sdis, rff, dmid, fmid, (float*)d_out);
}
